// SwinBlock_20109036880672
// MI455X (gfx1250) — matrix operand order repaired and checked by reading, not compiled, not run on hardware
//
#include <hip/hip_runtime.h>

// ---------------------------------------------------------------------------
// Types for CDNA5 WMMA (wave32, 16x16x32 bf16 -> f32)
// ---------------------------------------------------------------------------
typedef __bf16 bf16_t;
typedef __attribute__((ext_vector_type(16))) __bf16 v16bf;
typedef __attribute__((ext_vector_type(8)))  __bf16 v8bf;
typedef __attribute__((ext_vector_type(4)))  __bf16 v4bf;
typedef __attribute__((ext_vector_type(8)))  float  v8f;

#define DIM    192
#define C3     576
#define HEADS  6
#define HD     32
#define NWIN   64      // tokens per 8x8 window
#define IMG    128
#define SHIFTW 4
#define FF     768
#define QSCALE 0.17677669529663687f   // 32^-0.5
#define LNEPS  1e-5f

// A fragment: 16(M) x 32(K) bf16, row-major source with leading dim ld.
// Lane L (0-15): row M=L, elems 0..15 = K 0..15.
// Lane L+16:     row M=L, elems 0..15 = K 16..31.
__device__ __forceinline__ v16bf load_A_frag(const bf16_t* base, int ld) {
  const int lane = threadIdx.x & 31;
  const bf16_t* p = base + (lane & 15) * ld + ((lane >> 4) << 4);
  return *(const v16bf*)(p);
}

// B fragment: 32(K) x 16(N) bf16 sourced from a transposed [N][K] buffer with
// leading dim ld (= K stride). Lane n (0-15): col N=n, K = 0..15 contiguous;
// lane n+16: col N=n, K = 16..31 contiguous.
__device__ __forceinline__ v16bf load_B_frag(const bf16_t* baseT, int ld) {
  const int lane = threadIdx.x & 31;
  return *(const v16bf*)(baseT + (lane & 15) * ld + ((lane >> 4) << 4));
}

__device__ __forceinline__ v8f wmma_bf16(v16bf a, v16bf b, v8f c) {
  return __builtin_amdgcn_wmma_f32_16x16x32_bf16(false, a, false, b, (short)0, c,
                                                 false, false);
}

__device__ __forceinline__ v8f vzero8() {
  v8f z = {0.f, 0.f, 0.f, 0.f, 0.f, 0.f, 0.f, 0.f};
  return z;
}

// 16x(4x16) output macro-tile GEMM over KSTEPS k-slabs of 32, with register
// double-buffering: fragments for kb+1 are issued before the 4 WMMAs of kb so
// the global/LDS loads overlap the matrix pipe instead of draining loadcnt
// to zero in front of every WMMA.
template <int KSTEPS>
__device__ __forceinline__ void gemm4(const bf16_t* __restrict__ Abase, int lda,
                                      const bf16_t* __restrict__ Bbase, int ldb,
                                      v8f acc[4]) {
  v16bf a = load_A_frag(Abase, lda);
  v16bf b[4];
#pragma unroll
  for (int j = 0; j < 4; ++j) b[j] = load_B_frag(Bbase + j * 16 * ldb, ldb);
#pragma unroll
  for (int kb = 0; kb < KSTEPS - 1; ++kb) {
    v16bf an = load_A_frag(Abase + (kb + 1) * 32, lda);
    v16bf bn[4];
#pragma unroll
    for (int j = 0; j < 4; ++j)
      bn[j] = load_B_frag(Bbase + j * 16 * ldb + (kb + 1) * 32, ldb);
#pragma unroll
    for (int j = 0; j < 4; ++j) acc[j] = wmma_bf16(a, b[j], acc[j]);
    a = an;
#pragma unroll
    for (int j = 0; j < 4; ++j) b[j] = bn[j];
  }
#pragma unroll
  for (int j = 0; j < 4; ++j) acc[j] = wmma_bf16(a, b[j], acc[j]);
}

// ---------------------------------------------------------------------------
// Prep kernels
// ---------------------------------------------------------------------------
// in: [K][N] f32 row-major  ->  out: [N][K] bf16 row-major
__global__ void transpose_bf16_kernel(const float* __restrict__ in,
                                      bf16_t* __restrict__ out, int K, int N) {
  int i = blockIdx.x * 256 + threadIdx.x;
  if (i >= K * N) return;
  int n = i / K;
  int k = i - n * K;
  out[i] = (bf16_t)in[k * N + n];
}

// biasg[h][i][j] = bias_table[((ih-jh+7)*15 + (iw-jw+7))][h]
__global__ void bias_expand_kernel(const float* __restrict__ table,
                                   float* __restrict__ biasg) {
  int i = blockIdx.x * 256 + threadIdx.x;
  if (i >= HEADS * NWIN * NWIN) return;
  int h = i / (NWIN * NWIN);
  int r = (i / NWIN) % NWIN;
  int c = i % NWIN;
  int ih = r >> 3, iw = r & 7, jh = c >> 3, jw = c & 7;
  int idx = (ih - jh + 7) * 15 + (iw - jw + 7);
  biasg[i] = table[idx * HEADS + h];
}

// ---------------------------------------------------------------------------
// Fused window-attention kernel: LN1 + shift/partition + QKV + attention +
// proj + un-shift + residual.  One workgroup (256 threads = 8 waves) per
// window; 4096 windows.
// ---------------------------------------------------------------------------
__global__ __launch_bounds__(256) void swin_attn_kernel(
    const float* __restrict__ x, const float* __restrict__ gamma1,
    const float* __restrict__ beta1, const bf16_t* __restrict__ wqkvT,
    const float* __restrict__ bqkv, const float* __restrict__ biasg,
    const bf16_t* __restrict__ wprojT, const float* __restrict__ bproj,
    float* __restrict__ x1) {
  __shared__ __align__(128) char smem[139264];
  bf16_t* Xs = (bf16_t*)smem;          // [64][192]  LN'ed window tokens
  bf16_t* Qs = Xs + NWIN * DIM;        // [6][64][32] (pre-scaled)
  bf16_t* Ks = Qs + HEADS * NWIN * HD; // [6][64][32]
  bf16_t* Vt = Ks + HEADS * NWIN * HD; // [6][32][64] (transposed)
  bf16_t* Pw = Vt + HEADS * HD * NWIN; // [8 waves][16][64] softmax staging
  bf16_t* Os = Pw + 8 * 16 * NWIN;     // [64][192] attention output

  const int tid = threadIdx.x;
  const int bImg = blockIdx.x >> 8;
  const int wy = (blockIdx.x >> 4) & 15;
  const int wx = blockIdx.x & 15;

  // ---- Phase 0: LayerNorm + shifted gather into LDS (bf16) ----
  {
    const int t = tid >> 2;      // token 0..63
    const int prt = tid & 3;     // 48 channels each
    const int py = ((wy << 3) + (t >> 3) + SHIFTW) & (IMG - 1);
    const int px = ((wx << 3) + (t & 7) + SHIFTW) & (IMG - 1);
    const float* xr = x + (((bImg * IMG + py) * IMG + px) * DIM) + prt * 48;
    float4 va[12];
    float s = 0.f, ss = 0.f;
#pragma unroll
    for (int i = 0; i < 12; ++i) {
      va[i] = ((const float4*)xr)[i];
      s += va[i].x + va[i].y + va[i].z + va[i].w;
      ss += va[i].x * va[i].x + va[i].y * va[i].y + va[i].z * va[i].z +
            va[i].w * va[i].w;
    }
    s += __shfl_xor(s, 1);
    s += __shfl_xor(s, 2);
    ss += __shfl_xor(ss, 1);
    ss += __shfl_xor(ss, 2);
    const float mean = s * (1.0f / DIM);
    const float inv = rsqrtf(ss * (1.0f / DIM) - mean * mean + LNEPS);
#pragma unroll
    for (int i = 0; i < 12; ++i) {
      const float* v4 = (const float*)&va[i];
      const int c = prt * 48 + i * 4;
      v4bf o;
#pragma unroll
      for (int j = 0; j < 4; ++j)
        o[j] = (bf16_t)((v4[j] - mean) * inv * gamma1[c + j] + beta1[c + j]);
      *(v4bf*)(Xs + t * DIM + c) = o;   // packed ds_store_b64
    }
  }
  __syncthreads();

  const int wave = tid >> 5;
  const int lane = tid & 31;
  const int col = lane & 15;
  const int rhi = (lane >> 4) << 3;  // row offset 0 or 8 inside 16x16 C tile

  // ---- Phase 1: QKV GEMM  [64 x 576] = Xs[64x192] @ Wqkv, K=192 ----
  // 4 rowblocks x 9 groups of 4 column tiles; A fragment shared by 4 WMMAs.
  for (int g = wave; g < 36; g += 8) {
    const int rb = g / 9;
    const int cb0 = (g % 9) * 4;
    v8f acc[4] = {vzero8(), vzero8(), vzero8(), vzero8()};
    gemm4<6>(Xs + rb * 16 * DIM, DIM, wqkvT + cb0 * 16 * DIM, DIM, acc);
#pragma unroll
    for (int j = 0; j < 4; ++j) {
      const int cb = cb0 + j;
      const int which = cb / 12;            // 0=Q 1=K 2=V (tile-uniform)
      const int hh = (cb % 12) >> 1;        // head
      const int d = ((cb & 1) << 4) + col;  // 0..31
      const float bb = bqkv[cb * 16 + col];
      const float scl = (which == 0) ? QSCALE : 1.0f;
#pragma unroll
      for (int r = 0; r < 8; ++r) {
        const int row = rb * 16 + r + rhi;
        const float v = (acc[j][r] + bb) * scl;
        // Qs/Ks/Vt are contiguous: branch-free index select.
        const int idxQK = which * (HEADS * NWIN * HD) + (hh * NWIN + row) * HD + d;
        const int idxV = 2 * (HEADS * NWIN * HD) + (hh * HD + d) * NWIN + row;
        Qs[(which == 2) ? idxV : idxQK] = (bf16_t)v;
      }
    }
  }
  __syncthreads();

  // ---- Phase 2: attention. Task = (rowblock rb, head h); 24 tasks ----
  for (int task = wave; task < 24; task += 8) {
    const int rb = task & 3;
    const int h = task >> 2;
    // S = Q Kt : one WMMA per 16x16 tile (K = d = 32)
    v8f sacc[4];
    v16bf aq = load_A_frag(Qs + (h * NWIN + rb * 16) * HD, HD);
#pragma unroll
    for (int t = 0; t < 4; ++t) {
      v16bf bk = load_B_frag(Ks + (h * NWIN + t * 16) * HD, HD);
      sacc[t] = wmma_bf16(aq, bk, vzero8());
    }
    // bias + softmax per row (C layout: row r+rhi lives in 16-lane half)
    bf16_t* P = Pw + wave * 16 * NWIN;
#pragma unroll
    for (int r = 0; r < 8; ++r) {
      const int qrow = rb * 16 + r + rhi;
      float sv[4];
      float m = -3.0e38f;
#pragma unroll
      for (int t = 0; t < 4; ++t) {
        sv[t] = sacc[t][r] + biasg[(h * NWIN + qrow) * NWIN + t * 16 + col];
        m = fmaxf(m, sv[t]);
      }
      m = fmaxf(m, __shfl_xor(m, 1));
      m = fmaxf(m, __shfl_xor(m, 2));
      m = fmaxf(m, __shfl_xor(m, 4));
      m = fmaxf(m, __shfl_xor(m, 8));
      float sum = 0.f;
#pragma unroll
      for (int t = 0; t < 4; ++t) {
        sv[t] = __expf(sv[t] - m);
        sum += sv[t];
      }
      sum += __shfl_xor(sum, 1);
      sum += __shfl_xor(sum, 2);
      sum += __shfl_xor(sum, 4);
      sum += __shfl_xor(sum, 8);
      const float rs = 1.0f / sum;
#pragma unroll
      for (int t = 0; t < 4; ++t)
        P[(r + rhi) * NWIN + t * 16 + col] = (bf16_t)(sv[t] * rs);
    }
    // O = P V : [16 x 32], K = 64 (same-wave LDS is in-order, no barrier)
#pragma unroll
    for (int ct = 0; ct < 2; ++ct) {
      v8f acc = vzero8();
#pragma unroll
      for (int kb = 0; kb < 2; ++kb) {
        v16bf ap = load_A_frag(P + kb * 32, NWIN);
        v16bf bv = load_B_frag(Vt + (h * HD + ct * 16) * NWIN + kb * 32, NWIN);
        acc = wmma_bf16(ap, bv, acc);
      }
#pragma unroll
      for (int r = 0; r < 8; ++r)
        Os[(rb * 16 + r + rhi) * DIM + h * HD + ct * 16 + col] = (bf16_t)acc[r];
    }
  }
  __syncthreads();

  // ---- Phase 3: proj GEMM + un-shift + residual -> x1 ----
  // 4 rowblocks x 3 groups of 4 column tiles = 12 groups.
  for (int g = wave; g < 12; g += 8) {
    const int rb = g / 3;
    const int cb0 = (g % 3) * 4;
    v8f acc[4] = {vzero8(), vzero8(), vzero8(), vzero8()};
    gemm4<6>(Os + rb * 16 * DIM, DIM, wprojT + cb0 * 16 * DIM, DIM, acc);
#pragma unroll
    for (int j = 0; j < 4; ++j) {
      const int c = (cb0 + j) * 16 + col;
      const float bb = bproj[c];
#pragma unroll
      for (int r = 0; r < 8; ++r) {
        const int t = rb * 16 + r + rhi;
        const int py = ((wy << 3) + (t >> 3) + SHIFTW) & (IMG - 1);
        const int px = ((wx << 3) + (t & 7) + SHIFTW) & (IMG - 1);
        const int idx = ((bImg * IMG + py) * IMG + px) * DIM + c;
        x1[idx] = x[idx] + acc[j][r] + bb;
      }
    }
  }
}

// ---------------------------------------------------------------------------
// Fused MLP kernel: LN2 + FC1(192->768) + GELU + FC2(768->192) + residual.
// One workgroup per 64 tokens; 4096 blocks. Hidden activations live in LDS.
// ---------------------------------------------------------------------------
__global__ __launch_bounds__(256) void swin_mlp_kernel(
    const float* __restrict__ x1, const float* __restrict__ gamma2,
    const float* __restrict__ beta2, const bf16_t* __restrict__ w1T,
    const float* __restrict__ b1, const bf16_t* __restrict__ w2T,
    const float* __restrict__ b2, float* __restrict__ out) {
  __shared__ __align__(128) char smem[122880];
  bf16_t* Ms = (bf16_t*)smem;    // [64][192] LN2 output
  bf16_t* Hs = Ms + NWIN * DIM;  // [64][768] gelu(FC1)

  const int tid = threadIdx.x;
  const int base_tok = blockIdx.x * 64;

  // ---- LN2 ----
  {
    const int t = tid >> 2;
    const int prt = tid & 3;
    const float* xr = x1 + (base_tok + t) * DIM + prt * 48;
    float4 va[12];
    float s = 0.f, ss = 0.f;
#pragma unroll
    for (int i = 0; i < 12; ++i) {
      va[i] = ((const float4*)xr)[i];
      s += va[i].x + va[i].y + va[i].z + va[i].w;
      ss += va[i].x * va[i].x + va[i].y * va[i].y + va[i].z * va[i].z +
            va[i].w * va[i].w;
    }
    s += __shfl_xor(s, 1);
    s += __shfl_xor(s, 2);
    ss += __shfl_xor(ss, 1);
    ss += __shfl_xor(ss, 2);
    const float mean = s * (1.0f / DIM);
    const float inv = rsqrtf(ss * (1.0f / DIM) - mean * mean + LNEPS);
#pragma unroll
    for (int i = 0; i < 12; ++i) {
      const float* v4 = (const float*)&va[i];
      const int c = prt * 48 + i * 4;
      v4bf o;
#pragma unroll
      for (int j = 0; j < 4; ++j)
        o[j] = (bf16_t)((v4[j] - mean) * inv * gamma2[c + j] + beta2[c + j]);
      *(v4bf*)(Ms + t * DIM + c) = o;
    }
  }
  __syncthreads();

  const int wave = tid >> 5;
  const int lane = tid & 31;
  const int col = lane & 15;
  const int rhi = (lane >> 4) << 3;

  // ---- FC1 + GELU: [64 x 768], K = 192; 4 rb x 12 groups of 4 tiles ----
  for (int g = wave; g < 48; g += 8) {
    const int rb = g / 12;
    const int cb0 = (g % 12) * 4;
    v8f acc[4] = {vzero8(), vzero8(), vzero8(), vzero8()};
    gemm4<6>(Ms + rb * 16 * DIM, DIM, w1T + cb0 * 16 * DIM, DIM, acc);
#pragma unroll
    for (int j = 0; j < 4; ++j) {
      const int c = (cb0 + j) * 16 + col;
      const float bb = b1[c];
#pragma unroll
      for (int r = 0; r < 8; ++r) {
        const float v = acc[j][r] + bb;
        const float gl = 0.5f * v * (1.0f + erff(v * 0.70710678118654752f));
        Hs[(rb * 16 + r + rhi) * FF + c] = (bf16_t)gl;
      }
    }
  }
  __syncthreads();

  // ---- FC2 + residual: [64 x 192], K = 768; 4 rb x 3 groups of 4 tiles ----
  for (int g = wave; g < 12; g += 8) {
    const int rb = g / 3;
    const int cb0 = (g % 3) * 4;
    v8f acc[4] = {vzero8(), vzero8(), vzero8(), vzero8()};
    gemm4<24>(Hs + rb * 16 * FF, FF, w2T + cb0 * 16 * FF, FF, acc);
#pragma unroll
    for (int j = 0; j < 4; ++j) {
      const int c = (cb0 + j) * 16 + col;
      const float bb = b2[c];
#pragma unroll
      for (int r = 0; r < 8; ++r) {
        const int idx = (base_tok + rb * 16 + r + rhi) * DIM + c;
        out[idx] = x1[idx] + acc[j][r] + bb;
      }
    }
  }
}

// ---------------------------------------------------------------------------
// Launcher
// ---------------------------------------------------------------------------
extern "C" void kernel_launch(void* const* d_in, const int* in_sizes, int n_in,
                              void* d_out, int out_size, void* d_ws,
                              size_t ws_size, hipStream_t stream) {
  (void)in_sizes; (void)n_in; (void)out_size; (void)ws_size;
  const float* x      = (const float*)d_in[0];
  const float* gamma1 = (const float*)d_in[1];
  const float* beta1  = (const float*)d_in[2];
  const float* w_qkv  = (const float*)d_in[3];
  const float* b_qkv  = (const float*)d_in[4];
  const float* btab   = (const float*)d_in[5];
  const float* w_proj = (const float*)d_in[6];
  const float* b_proj = (const float*)d_in[7];
  const float* gamma2 = (const float*)d_in[8];
  const float* beta2  = (const float*)d_in[9];
  const float* w1     = (const float*)d_in[10];
  const float* b1     = (const float*)d_in[11];
  const float* w2     = (const float*)d_in[12];
  const float* b2     = (const float*)d_in[13];
  float* out = (float*)d_out;

  char* ws = (char*)d_ws;
  size_t off = 0;
  auto carve = [&](size_t bytes) {
    char* p = ws + off;
    off = (off + bytes + 255) & ~(size_t)255;
    return p;
  };
  float*  x1     = (float*)carve((size_t)16 * 128 * 128 * DIM * sizeof(float));
  bf16_t* wqkvT  = (bf16_t*)carve((size_t)C3 * DIM * sizeof(bf16_t));
  bf16_t* wprojT = (bf16_t*)carve((size_t)DIM * DIM * sizeof(bf16_t));
  bf16_t* w1T    = (bf16_t*)carve((size_t)FF * DIM * sizeof(bf16_t));
  bf16_t* w2T    = (bf16_t*)carve((size_t)DIM * FF * sizeof(bf16_t));
  float*  biasg  = (float*)carve((size_t)HEADS * NWIN * NWIN * sizeof(float));

  transpose_bf16_kernel<<<(DIM * C3 + 255) / 256, 256, 0, stream>>>(
      w_qkv, wqkvT, DIM, C3);
  transpose_bf16_kernel<<<(DIM * DIM + 255) / 256, 256, 0, stream>>>(
      w_proj, wprojT, DIM, DIM);
  transpose_bf16_kernel<<<(DIM * FF + 255) / 256, 256, 0, stream>>>(
      w1, w1T, DIM, FF);
  transpose_bf16_kernel<<<(FF * DIM + 255) / 256, 256, 0, stream>>>(
      w2, w2T, FF, DIM);
  bias_expand_kernel<<<(HEADS * NWIN * NWIN + 255) / 256, 256, 0, stream>>>(
      btab, biasg);

  swin_attn_kernel<<<4096, 256, 0, stream>>>(x, gamma1, beta1, wqkvT, b_qkv,
                                             biasg, wprojT, b_proj, x1);
  swin_mlp_kernel<<<4096, 256, 0, stream>>>(x1, gamma2, beta2, w1T, b1, w2T,
                                            b2, out);
}
